// Seq2SeqSum_37881611550875
// MI455X (gfx1250) — compile-verified, matrix operand order checked by reading, not run on hardware
//
#include <hip/hip_runtime.h>
#include <hip/hip_bf16.h>
#include <math.h>

// ---------------------------------------------------------------------------
// Seq2Seq LSTM (encoder biLSTM x3, decoder 3-cell + attention) for gfx1250.
// All GEMMs use v_wmma_f32_16x16x32_f16 (f16 in, f32 accum).
// Wave tile 32(M) x 64(N): 2 A-frags x 4 B-frags -> 8 WMMAs per 32-wide K step
// (8 accumulator chains for WMMA latency hiding, ~21 FLOP per byte of L0/L2
// operand traffic). Block = 4 waves side-by-side in N -> block tile 32 x 256.
// ---------------------------------------------------------------------------

typedef __attribute__((ext_vector_type(16))) _Float16 v16h;
typedef __attribute__((ext_vector_type(8)))  float    v8f;
typedef _Float16 half_t;

#define BB 32      // batch
#define LL 128     // src len
#define TT 64      // tgt len
#define EE 256     // embed dim
#define HH 512     // hidden
#define VV 32000   // vocab
#define G4H 2048   // 4*H

// ------------------------------- device helpers ----------------------------

__device__ __forceinline__ float sigmoidf_(float x) { return 1.0f / (1.0f + expf(-x)); }

// Load a 16x32 f16 tile fragment (row-major, leading dim ld halves).
// gfx1250 16-bit A/B layout: lane L holds row (L&15); elements 0..7 are K =
// k0..k0+7, elements 8..15 are K = k0+16..k0+23, where k0 = (L&16)>>1.
__device__ __forceinline__ v16h frag_load(const half_t* p, int ld, int lane) {
    int r  = lane & 15;
    int k0 = (lane & 16) >> 1;              // 0 or 8
    const half_t* q = p + r * ld + k0;
    union { v16h v; int4 x[2]; } u;
    u.x[0] = *reinterpret_cast<const int4*>(q);        // K = k0 .. k0+7
    u.x[1] = *reinterpret_cast<const int4*>(q + 16);   // K = k0+16 .. k0+23
    return u.v;
}

// C/D tile layout: element v of lane L is (M = v + ((L&16)>>1), N = L&15).
__device__ __forceinline__ v8f cload(const float* p, int ld, int lane) {
    int n  = lane & 15;
    int m0 = (lane & 16) >> 1;
    v8f c;
#pragma unroll
    for (int v = 0; v < 8; ++v) c[v] = p[(m0 + v) * ld + n];
    return c;
}

// ------------------------------- GEMM kernel -------------------------------
// C[M,N](f32) = act( A[M,K](f16,row-major) @ B[N,K](f16,row-major)^T
//                    + bias[N]? + Cacc? )
// Requires: M % 32 == 0, N % 256 == 0, K % 32 == 0 (true for all call sites).
__global__ __launch_bounds__(128) void gemm_wmma_kernel(
    const half_t* __restrict__ A, const half_t* __restrict__ Bm,
    const float* __restrict__ bias, const float* Cacc,
    float* C, half_t* Ch,
    int M, int N, int K, int act, int permT, int Bdim)
{
    const int lane = threadIdx.x & 31;
    const int wave = threadIdx.x >> 5;
    const int row0 = blockIdx.y * 32;
    const int col0 = blockIdx.x * 256 + wave * 64;
    if (row0 >= M) return;                  // wave-uniform

    v8f acc[2][4];
    if (Cacc) {
#pragma unroll
        for (int mi = 0; mi < 2; ++mi)
#pragma unroll
            for (int ni = 0; ni < 4; ++ni)
                acc[mi][ni] = cload(Cacc + (row0 + mi * 16) * N + col0 + ni * 16, N, lane);
    } else {
#pragma unroll
        for (int mi = 0; mi < 2; ++mi)
#pragma unroll
            for (int ni = 0; ni < 4; ++ni)
                acc[mi][ni] = (v8f){0.f,0.f,0.f,0.f,0.f,0.f,0.f,0.f};
    }

    const half_t* Ap = A  + (long)row0 * K;
    const half_t* Bp = Bm + (long)col0 * K;
    for (int k = 0; k < K; k += 32) {
        v16h a0 = frag_load(Ap + k,          K, lane);
        v16h a1 = frag_load(Ap + 16 * K + k, K, lane);
        v16h b0 = frag_load(Bp + k,          K, lane);
        v16h b1 = frag_load(Bp + 16 * K + k, K, lane);
        v16h b2 = frag_load(Bp + 32 * K + k, K, lane);
        v16h b3 = frag_load(Bp + 48 * K + k, K, lane);
        acc[0][0] = __builtin_amdgcn_wmma_f32_16x16x32_f16(false, a0, false, b0, (short)0, acc[0][0], false, false);
        acc[1][0] = __builtin_amdgcn_wmma_f32_16x16x32_f16(false, a1, false, b0, (short)0, acc[1][0], false, false);
        acc[0][1] = __builtin_amdgcn_wmma_f32_16x16x32_f16(false, a0, false, b1, (short)0, acc[0][1], false, false);
        acc[1][1] = __builtin_amdgcn_wmma_f32_16x16x32_f16(false, a1, false, b1, (short)0, acc[1][1], false, false);
        acc[0][2] = __builtin_amdgcn_wmma_f32_16x16x32_f16(false, a0, false, b2, (short)0, acc[0][2], false, false);
        acc[1][2] = __builtin_amdgcn_wmma_f32_16x16x32_f16(false, a1, false, b2, (short)0, acc[1][2], false, false);
        acc[0][3] = __builtin_amdgcn_wmma_f32_16x16x32_f16(false, a0, false, b3, (short)0, acc[0][3], false, false);
        acc[1][3] = __builtin_amdgcn_wmma_f32_16x16x32_f16(false, a1, false, b3, (short)0, acc[1][3], false, false);
    }

    const int nlo = lane & 15;
    const int m0  = (lane & 16) >> 1;
#pragma unroll
    for (int mi = 0; mi < 2; ++mi) {
#pragma unroll
        for (int ni = 0; ni < 4; ++ni) {
            const int n  = col0 + ni * 16 + nlo;
            const float bv = bias ? bias[n] : 0.0f;
            const int mb = row0 + mi * 16 + m0;
#pragma unroll
            for (int v = 0; v < 8; ++v) {
                const int m = mb + v;
                float x = acc[mi][ni][v] + bv;
                if (act == 1) x = tanhf(x);
                if (C) {
                    if (permT > 0) {            // row m = t*Bdim+b -> [b][t][n]
                        int t = m / Bdim, b = m % Bdim;
                        C[((long)b * permT + t) * (long)N + n] = x;
                    } else {
                        C[(long)m * N + n] = x;
                    }
                }
                if (Ch) Ch[(long)m * N + n] = (half_t)x;
            }
        }
    }
}

// ------------------------------- LSTM gates --------------------------------
// g: [32, 2048] pre-activation gates (i,f,g,o chunks of 512). Updates h,c
// (f32 [32,512]) and writes f16 copies of h to up to two destinations.
__global__ __launch_bounds__(256) void lstm_gate_kernel(
    const float* __restrict__ g, float* h, float* c,
    half_t* d1, int ld1, half_t* d2, int ld2)
{
    int idx = blockIdx.x * blockDim.x + threadIdx.x;
    if (idx >= BB * HH) return;
    int b = idx >> 9, j = idx & (HH - 1);
    const float* gr = g + b * G4H;
    float gi = gr[j], gf = gr[j + HH], gg = gr[j + 2 * HH], go = gr[j + 3 * HH];
    float cn = sigmoidf_(gf) * c[idx] + sigmoidf_(gi) * tanhf(gg);
    float hn = sigmoidf_(go) * tanhf(cn);
    c[idx] = cn; h[idx] = hn;
    half_t hh = (half_t)hn;
    if (d1) d1[b * ld1 + j] = hh;
    if (d2) d2[b * ld2 + j] = hh;
}

// Broadcast [512] init vectors into [32,512] states.
__global__ __launch_bounds__(256) void bcast2_kernel(
    const float* __restrict__ hv, const float* __restrict__ cv, float* h, float* c)
{
    int idx = blockIdx.x * blockDim.x + threadIdx.x;
    if (idx >= BB * HH) return;
    int j = idx & (HH - 1);
    h[idx] = hv[j]; c[idx] = cv[j];
}

// f32 [32,512] -> f16, up to two strided destinations.
__global__ __launch_bounds__(256) void scatter16_kernel(
    const float* __restrict__ src, half_t* d1, int ld1, half_t* d2, int ld2)
{
    int idx = blockIdx.x * blockDim.x + threadIdx.x;
    if (idx >= BB * HH) return;
    int b = idx >> 9, j = idx & (HH - 1);
    half_t v = (half_t)src[idx];
    if (d1) d1[b * ld1 + j] = v;
    if (d2) d2[b * ld2 + j] = v;
}

// Final encoder h,c -> f16 concat buffers (cols dir*512 of [32,1024]).
__global__ __launch_bounds__(256) void scatterHC_kernel(
    const float* __restrict__ h, const float* __restrict__ c,
    half_t* hd, half_t* cd, int ld)
{
    int idx = blockIdx.x * blockDim.x + threadIdx.x;
    if (idx >= BB * HH) return;
    int b = idx >> 9, j = idx & (HH - 1);
    hd[b * ld + j] = (half_t)h[idx];
    cd[b * ld + j] = (half_t)c[idx];
}

// f32 [rows,cols] -> f16 packed into dst with leading dim ldd at column off.
__global__ __launch_bounds__(256) void convert_pack_kernel(
    const float* __restrict__ src, half_t* __restrict__ dst,
    long n, int cols, int ldd, int off)
{
    long i = (long)blockIdx.x * blockDim.x + threadIdx.x;
    if (i >= n) return;
    long r = i / cols; int k = (int)(i - r * cols);
    dst[r * ldd + off + k] = (half_t)src[i];
}

// Embedding gather: tokens [B, Lseq] -> out row r = t*B + b (f16, 256 cols).
__global__ __launch_bounds__(256) void embed_kernel(
    const int* __restrict__ tok, const float* __restrict__ emb,
    half_t* __restrict__ out, int Lseq)
{
    long i = (long)blockIdx.x * blockDim.x + threadIdx.x;
    long n = (long)Lseq * BB * EE;
    if (i >= n) return;
    int e = (int)(i & (EE - 1));
    int r = (int)(i >> 8);               // t*BB + b
    int t = r / BB, b = r % BB;
    int id = tok[b * Lseq + t];
    out[i] = (half_t)emb[(long)id * EE + e];
}

// Attention scores + masked softmax. Block b (128 threads, one per l).
__global__ __launch_bounds__(128) void attn_softmax_kernel(
    const float* __restrict__ h3, const float* __restrict__ keys,
    const int* __restrict__ lens, float* __restrict__ alpha)
{
    __shared__ float red[LL];
    const int b = blockIdx.x, l = threadIdx.x;
    const float* hb = h3 + b * HH;
    const float* kb = keys + ((long)l * BB + b) * HH;
    float s = 0.f;
    for (int j = 0; j < HH; ++j) s += hb[j] * kb[j];
    if (l >= lens[b]) s = -1e18f;
    red[l] = s; __syncthreads();
    for (int o = 64; o > 0; o >>= 1) { if (l < o) red[l] = fmaxf(red[l], red[l + o]); __syncthreads(); }
    float mx = red[0]; __syncthreads();
    float e = expf(s - mx);
    red[l] = e; __syncthreads();
    for (int o = 64; o > 0; o >>= 1) { if (l < o) red[l] += red[l + o]; __syncthreads(); }
    alpha[b * LL + l] = e / red[0];
}

// ctx[b,d] = sum_l alpha[b,l] * enc_outs[l,b,d]; writes f16 into ctxh3[:,0:1024].
__global__ __launch_bounds__(256) void ctx_kernel(
    const float* __restrict__ alpha, const half_t* __restrict__ encOuts,
    half_t* __restrict__ ctxh3)
{
    int idx = blockIdx.x * blockDim.x + threadIdx.x;
    if (idx >= BB * 2 * HH) return;
    int b = idx >> 10, d = idx & (2 * HH - 1);
    const float* ab = alpha + b * LL;
    float s = 0.f;
    for (int l = 0; l < LL; ++l)
        s += ab[l] * (float)encOuts[((long)l * BB + b) * (2 * HH) + d];
    ctxh3[b * (3 * HH) + d] = (half_t)s;
}

// In-place log-softmax over each V-length row of d_out. Block per row.
__global__ __launch_bounds__(256) void logsoftmax_kernel(float* out)
{
    __shared__ float red[256];
    float* p = out + (long)blockIdx.x * VV;
    const int tid = threadIdx.x;
    float m = -1e30f;
    for (int i = tid; i < VV; i += 256) m = fmaxf(m, p[i]);
    red[tid] = m; __syncthreads();
    for (int o = 128; o > 0; o >>= 1) { if (tid < o) red[tid] = fmaxf(red[tid], red[tid + o]); __syncthreads(); }
    m = red[0]; __syncthreads();
    float s = 0.f;
    for (int i = tid; i < VV; i += 256) s += expf(p[i] - m);
    red[tid] = s; __syncthreads();
    for (int o = 128; o > 0; o >>= 1) { if (tid < o) red[tid] += red[tid + o]; __syncthreads(); }
    const float lse = m + logf(red[0]);
    __syncthreads();
    for (int i = tid; i < VV; i += 256) p[i] -= lse;
}

// ------------------------------- host side ---------------------------------

namespace {

struct Ins {
    const float *attn_W, *attn_b;
    const float *cW[3], *cU[3], *cb[3];
    const float *concat_W, *concat_b, *dec_c_W, *dec_h_W, *emb;
    const float *enc_init_c, *enc_init_h;
    const float *encW[3][2], *encU[3][2], *encB[3][2];
    const float *out_W, *out_b;
    const int *src, *src_lengths, *tgt;
};

struct Bump { char* base; size_t off; };
template <typename T> T* wsA(Bump& w, size_t elems) {
    size_t bytes = (elems * sizeof(T) + 255) & ~size_t(255);
    T* p = (T*)(w.base + w.off); w.off += bytes; return p;
}

inline void gemm(hipStream_t s, const half_t* A, const half_t* B, const float* bias,
                 const float* Cacc, float* C, half_t* Ch,
                 int M, int N, int K, int act = 0, int permT = 0, int Bdim = 0) {
    dim3 g(N / 256, M / 32);
    gemm_wmma_kernel<<<g, 128, 0, s>>>(A, B, bias, Cacc, C, Ch, M, N, K, act, permT, Bdim);
}
inline void cvt(hipStream_t s, const float* src, half_t* dst, long n, int cols, int ldd, int off) {
    long blocks = (n + 255) / 256;
    convert_pack_kernel<<<(unsigned)blocks, 256, 0, s>>>(src, dst, n, cols, ldd, off);
}

} // namespace

extern "C" void kernel_launch(void* const* d_in, const int* in_sizes, int n_in,
                              void* d_out, int out_size, void* d_ws, size_t ws_size,
                              hipStream_t stream)
{
    Ins in;
    const float* const* F = (const float* const*)d_in;
    if (in_sizes[0] == BB * LL && in_sizes[2] == BB * TT) {
        // insertion order: src, src_lengths, tgt, then params in declaration order
        in.src = (const int*)d_in[0]; in.src_lengths = (const int*)d_in[1]; in.tgt = (const int*)d_in[2];
        in.emb = F[3];
        int i = 4;
        for (int l = 0; l < 3; ++l) {
            in.encW[l][0] = F[i++]; in.encU[l][0] = F[i++]; in.encB[l][0] = F[i++];
            in.encW[l][1] = F[i++]; in.encU[l][1] = F[i++]; in.encB[l][1] = F[i++];
        }
        in.enc_init_h = F[i++]; in.enc_init_c = F[i++];
        in.dec_h_W = F[i++]; in.dec_c_W = F[i++];
        for (int c = 0; c < 3; ++c) { in.cW[c] = F[i++]; in.cU[c] = F[i++]; in.cb[c] = F[i++]; }
        in.attn_W = F[i++]; in.attn_b = F[i++];
        in.concat_W = F[i++]; in.concat_b = F[i++];
        in.out_W = F[i++]; in.out_b = F[i++];
    } else {
        // jax-pytree sorted-key order
        int i = 0;
        in.attn_W = F[i++]; in.attn_b = F[i++];
        for (int c = 0; c < 3; ++c) { in.cW[c] = F[i++]; in.cU[c] = F[i++]; in.cb[c] = F[i++]; }
        in.concat_W = F[i++]; in.concat_b = F[i++];
        in.dec_c_W = F[i++]; in.dec_h_W = F[i++];
        in.emb = F[i++];
        in.enc_init_c = F[i++]; in.enc_init_h = F[i++];
        for (int l = 0; l < 3; ++l) {
            in.encW[l][0] = F[i++]; in.encU[l][0] = F[i++]; in.encB[l][0] = F[i++];
            in.encW[l][1] = F[i++]; in.encU[l][1] = F[i++]; in.encB[l][1] = F[i++];
        }
        in.out_W = F[i++]; in.out_b = F[i++];
        in.src = (const int*)d_in[i]; in.src_lengths = (const int*)d_in[i + 1]; in.tgt = (const int*)d_in[i + 2];
    }

    // ---------------- workspace layout ----------------
    Bump w{(char*)d_ws, 0};
    half_t* wEnc[3][2]; half_t* uEnc[3][2];
    for (int l = 0; l < 3; ++l) {
        int din = (l == 0) ? EE : 2 * HH;
        for (int d = 0; d < 2; ++d) {
            wEnc[l][d] = wsA<half_t>(w, (size_t)G4H * din);
            uEnc[l][d] = wsA<half_t>(w, (size_t)G4H * HH);
        }
    }
    half_t* wC1W   = wsA<half_t>(w, (size_t)G4H * EE);
    half_t* wC1U   = wsA<half_t>(w, (size_t)G4H * HH);
    half_t* wW2cat = wsA<half_t>(w, (size_t)G4H * 2 * HH);
    half_t* wW3cat = wsA<half_t>(w, (size_t)G4H * 2 * HH);
    half_t* wAttn  = wsA<half_t>(w, (size_t)HH * 2 * HH);
    half_t* wConcat= wsA<half_t>(w, (size_t)HH * 3 * HH);
    half_t* wDecH  = wsA<half_t>(w, (size_t)HH * 2 * HH);
    half_t* wDecC  = wsA<half_t>(w, (size_t)HH * 2 * HH);
    half_t* wOut   = wsA<half_t>(w, (size_t)VV * HH);
    half_t* xA     = wsA<half_t>(w, (size_t)LL * BB * 2 * HH);
    half_t* xB     = wsA<half_t>(w, (size_t)LL * BB * 2 * HH);
    half_t* tgtEmb = wsA<half_t>(w, (size_t)TT * BB * EE);
    half_t* hstate16 = wsA<half_t>(w, (size_t)BB * HH);
    half_t* h1_16  = wsA<half_t>(w, (size_t)BB * HH);
    half_t* cat12  = wsA<half_t>(w, (size_t)BB * 2 * HH);
    half_t* cat23  = wsA<half_t>(w, (size_t)BB * 2 * HH);
    half_t* ctxh3  = wsA<half_t>(w, (size_t)BB * 3 * HH);
    half_t* hcat[3]; half_t* ccat[3];
    for (int l = 0; l < 3; ++l) { hcat[l] = wsA<half_t>(w, (size_t)BB * 2 * HH);
                                  ccat[l] = wsA<half_t>(w, (size_t)BB * 2 * HH); }
    half_t* ccAll  = wsA<half_t>(w, (size_t)TT * BB * HH);
    float* Xgbuf   = wsA<float>(w, (size_t)LL * BB * G4H);
    float* XgDec1  = wsA<float>(w, (size_t)TT * BB * G4H);
    float* g2buf   = wsA<float>(w, (size_t)BB * G4H);
    float* g3buf   = wsA<float>(w, (size_t)BB * G4H);
    float* hF      = wsA<float>(w, (size_t)BB * HH);
    float* cF      = wsA<float>(w, (size_t)BB * HH);
    float* keys    = wsA<float>(w, (size_t)LL * BB * HH);
    float* alpha   = wsA<float>(w, (size_t)BB * LL);
    float* dh[3]; float* dc[3];
    for (int l = 0; l < 3; ++l) { dh[l] = wsA<float>(w, (size_t)BB * HH);
                                  dc[l] = wsA<float>(w, (size_t)BB * HH); }
    (void)ws_size; (void)n_in; (void)out_size;

    // ---------------- weight conversion (f32 -> f16) ----------------
    for (int l = 0; l < 3; ++l) {
        int din = (l == 0) ? EE : 2 * HH;
        for (int d = 0; d < 2; ++d) {
            cvt(stream, in.encW[l][d], wEnc[l][d], (long)G4H * din, din, din, 0);
            cvt(stream, in.encU[l][d], uEnc[l][d], (long)G4H * HH, HH, HH, 0);
        }
    }
    cvt(stream, in.cW[0], wC1W, (long)G4H * EE, EE, EE, 0);
    cvt(stream, in.cU[0], wC1U, (long)G4H * HH, HH, HH, 0);
    cvt(stream, in.cW[1], wW2cat, (long)G4H * HH, HH, 2 * HH, 0);
    cvt(stream, in.cU[1], wW2cat, (long)G4H * HH, HH, 2 * HH, HH);
    cvt(stream, in.cW[2], wW3cat, (long)G4H * HH, HH, 2 * HH, 0);
    cvt(stream, in.cU[2], wW3cat, (long)G4H * HH, HH, 2 * HH, HH);
    cvt(stream, in.attn_W,   wAttn,   (long)HH * 2 * HH, 2 * HH, 2 * HH, 0);
    cvt(stream, in.concat_W, wConcat, (long)HH * 3 * HH, 3 * HH, 3 * HH, 0);
    cvt(stream, in.dec_h_W,  wDecH,   (long)HH * 2 * HH, 2 * HH, 2 * HH, 0);
    cvt(stream, in.dec_c_W,  wDecC,   (long)HH * 2 * HH, 2 * HH, 2 * HH, 0);
    cvt(stream, in.out_W,    wOut,    (long)VV * HH, HH, HH, 0);

    // ---------------- embeddings ----------------
    embed_kernel<<<((long)LL * BB * EE + 255) / 256, 256, 0, stream>>>(in.src, in.emb, xA, LL);
    embed_kernel<<<((long)TT * BB * EE + 255) / 256, 256, 0, stream>>>(in.tgt, in.emb, tgtEmb, TT);

    const int SB = (BB * HH + 255) / 256;   // 64 blocks for [32,512] elementwise

    // ---------------- encoder: 3 bidirectional layers ----------------
    half_t* xcur = xA; half_t* xnext = xB; int ldx = EE;
    for (int l = 0; l < 3; ++l) {
        for (int d = 0; d < 2; ++d) {
            // time-parallel input projection (+bias): Xg = X @ W^T + b
            gemm(stream, xcur, wEnc[l][d], in.encB[l][d], nullptr, Xgbuf, nullptr,
                 LL * BB, G4H, ldx);
            bcast2_kernel<<<SB, 256, 0, stream>>>(in.enc_init_h + (2 * l + d) * HH,
                                                  in.enc_init_c + (2 * l + d) * HH, hF, cF);
            scatter16_kernel<<<SB, 256, 0, stream>>>(hF, hstate16, HH, nullptr, 0);
            for (int step = 0; step < LL; ++step) {
                int t = d ? (LL - 1 - step) : step;
                float* gslice = Xgbuf + (size_t)t * BB * G4H;
                gemm(stream, hstate16, uEnc[l][d], nullptr, gslice, gslice, nullptr,
                     BB, G4H, HH);
                lstm_gate_kernel<<<SB, 256, 0, stream>>>(gslice, hF, cF,
                    xnext + (size_t)t * BB * 2 * HH + d * HH, 2 * HH,
                    hstate16, HH);
            }
            scatterHC_kernel<<<SB, 256, 0, stream>>>(hF, cF, hcat[l] + d * HH,
                                                     ccat[l] + d * HH, 2 * HH);
        }
        half_t* tmp = xcur; xcur = xnext; xnext = tmp; ldx = 2 * HH;
    }
    half_t* encOuts = xcur;   // [L*B, 2H] f16, row = l*B + b

    // attention keys: [L*B, H] = encOuts @ attn_W^T + attn_b
    gemm(stream, encOuts, wAttn, in.attn_b, nullptr, keys, nullptr, LL * BB, HH, 2 * HH);

    // decoder init states: h_l = cat(hTf,hTb) @ dec_h_W^T ; c_l likewise
    for (int l = 0; l < 3; ++l) {
        gemm(stream, hcat[l], wDecH, nullptr, nullptr, dh[l], nullptr, BB, HH, 2 * HH);
        gemm(stream, ccat[l], wDecC, nullptr, nullptr, dc[l], nullptr, BB, HH, 2 * HH);
    }
    scatter16_kernel<<<SB, 256, 0, stream>>>(dh[0], h1_16, HH, nullptr, 0);
    scatter16_kernel<<<SB, 256, 0, stream>>>(dh[1], cat12 + HH, 2 * HH, nullptr, 0);
    scatter16_kernel<<<SB, 256, 0, stream>>>(dh[2], cat23 + HH, 2 * HH, ctxh3 + 2 * HH, 3 * HH);

    // decoder cell-1 input gates for all steps: XgDec1 = tgtEmb @ c1W^T + c1b
    gemm(stream, tgtEmb, wC1W, in.cb[0], nullptr, XgDec1, nullptr, TT * BB, G4H, EE);

    // ---------------- decoder time loop ----------------
    for (int t = 0; t < TT; ++t) {
        float* g1 = XgDec1 + (size_t)t * BB * G4H;
        gemm(stream, h1_16, wC1U, nullptr, g1, g1, nullptr, BB, G4H, HH);
        lstm_gate_kernel<<<SB, 256, 0, stream>>>(g1, dh[0], dc[0],
                                                 h1_16, HH, cat12, 2 * HH);
        gemm(stream, cat12, wW2cat, in.cb[1], nullptr, g2buf, nullptr, BB, G4H, 2 * HH);
        lstm_gate_kernel<<<SB, 256, 0, stream>>>(g2buf, dh[1], dc[1],
                                                 cat12 + HH, 2 * HH, cat23, 2 * HH);
        gemm(stream, cat23, wW3cat, in.cb[2], nullptr, g3buf, nullptr, BB, G4H, 2 * HH);
        lstm_gate_kernel<<<SB, 256, 0, stream>>>(g3buf, dh[2], dc[2],
                                                 cat23 + HH, 2 * HH, ctxh3 + 2 * HH, 3 * HH);
        attn_softmax_kernel<<<BB, LL, 0, stream>>>(dh[2], keys, in.src_lengths, alpha);
        ctx_kernel<<<(BB * 2 * HH + 255) / 256, 256, 0, stream>>>(alpha, encOuts, ctxh3);
        gemm(stream, ctxh3, wConcat, in.concat_b, nullptr, nullptr,
             ccAll + (size_t)t * BB * HH, BB, HH, 3 * HH, /*act=tanh*/1);
    }

    // ---------------- output projection + log-softmax ----------------
    // logits[t*B+b, :] -> d_out[b][t][:], then in-place log-softmax per row.
    gemm(stream, ccAll, wOut, in.out_b, nullptr, (float*)d_out, nullptr,
         TT * BB, VV, HH, 0, /*permT=*/TT, /*Bdim=*/BB);
    logsoftmax_kernel<<<TT * BB, 256, 0, stream>>>((float*)d_out);
}
